// LocalSelfAttention_opt_687194768281
// MI455X (gfx1250) — matrix operand order repaired and checked by reading, not compiled, not run on hardware
//
#include <hip/hip_runtime.h>
#include <cstdint>
#include <cstddef>

// ---------------------------------------------------------------------------
// Types for CDNA5 WMMA bf16 (wave32): A/B = 16 bf16 per lane, C/D = 8 f32.
// ---------------------------------------------------------------------------
typedef __attribute__((ext_vector_type(8)))  float  v8f;
typedef __attribute__((ext_vector_type(8)))  __bf16 bf16x8;
typedef __attribute__((ext_vector_type(16))) __bf16 bf16x16;

union ABFrag { bf16x16 v; bf16x8 h[2]; };

__device__ __forceinline__ float bf2f(unsigned short u) {
  union { unsigned int i; float f; } c; c.i = ((unsigned int)u) << 16; return c.f;
}
__device__ __forceinline__ unsigned short f2bf(float f) {
  union { float f; unsigned int i; } c; c.f = f;
  unsigned int u = c.i;
  u += 0x7FFFu + ((u >> 16) & 1u);          // round-to-nearest-even
  return (unsigned short)(u >> 16);
}

// ---------------------------------------------------------------------------
// Problem constants
// ---------------------------------------------------------------------------
constexpr int KDIM   = 512;                 // D = HEADS*DK = 512 (both GEMMs K)
constexpr int S      = 256;
constexpr int HEADS  = 8;
constexpr int DK     = 64;
constexpr int KSIZE  = 7;                   // 1 + 2*3 neighbors
constexpr int MROWS  = 2 * 64 * 256;        // B*HW*S = 32768
constexpr int NQKV   = 3 * HEADS * DK;      // 1536

// ---------------------------------------------------------------------------
// Elementwise fp32 -> bf16 (vectorized by 4)
// ---------------------------------------------------------------------------
__global__ __launch_bounds__(256) void k_cvt_x(const float* __restrict__ X,
                                               unsigned short* __restrict__ O,
                                               int n4) {
  int i = blockIdx.x * blockDim.x + threadIdx.x;
  if (i >= n4) return;
  const float4 v = ((const float4*)X)[i];
  uint2 o;
  o.x = (unsigned)f2bf(v.x) | ((unsigned)f2bf(v.y) << 16);
  o.y = (unsigned)f2bf(v.z) | ((unsigned)f2bf(v.w) << 16);
  ((uint2*)O)[i] = o;
}

// ---------------------------------------------------------------------------
// Transpose + convert weights:  W [512][N] f32  ->  WT [N][512] bf16.
// Columns n < scaleCols get * 0.125 (fold 1/sqrt(DK) into the q-projection).
// ---------------------------------------------------------------------------
__global__ __launch_bounds__(256) void k_prep_w(const float* __restrict__ W,
                                                unsigned short* __restrict__ WT,
                                                int N, int scaleCols) {
  int idx = blockIdx.x * blockDim.x + threadIdx.x;
  if (idx >= N * KDIM) return;
  int k = idx & (KDIM - 1);
  int n = idx >> 9;                          // KDIM == 512
  float v = W[(size_t)k * N + n];
  if (n < scaleCols) v *= 0.125f;
  WT[idx] = f2bf(v);
}

// ---------------------------------------------------------------------------
// bf16 WMMA GEMM:  C[M][ldc] = A[M][512] * BT[N][512]^T
// Block = 256 threads = 8 waves; block tile 128(M) x 128(N);
// wave tile 32(M) x 64(N) = 2x4 WMMA 16x16 tiles, K-loop step 32.
//
// Fragment packing per ISA (16-bit A 16x32): lane<16 -> K {0..7,16..23},
// lane>=16 -> K {8..15,24..31}; B rows taken from transposed weights and packed with
// the same K order as A (lane<16: K {0..7,16..23}, lane>=16: K {8..15,24..31}).
// ---------------------------------------------------------------------------
template <bool F32OUT>
__global__ __launch_bounds__(256) void gemm_bf16_wmma(
    const unsigned short* __restrict__ A,    // [M][512] bf16 row-major
    const unsigned short* __restrict__ BT,   // [N][512] bf16 (weights^T)
    void* __restrict__ Cout, int ldc) {
  const int wid  = threadIdx.x >> 5;
  const int lane = threadIdx.x & 31;
  const int half = lane >> 4;
  const int l16  = lane & 15;

  const int tileN0 = blockIdx.x * 128 + (wid & 1) * 64;
  const int tileM0 = blockIdx.y * 128 + (wid >> 1) * 32;

  v8f acc[2][4] = {};

  const unsigned short* Ar0 = A  + (size_t)(tileM0 + l16) * KDIM;
  const unsigned short* Ar1 = Ar0 + (size_t)16 * KDIM;
  const unsigned short* Br0 = BT + (size_t)(tileN0 + l16) * KDIM;

#pragma unroll 4
  for (int k = 0; k < KDIM; k += 32) {
    ABFrag a[2], b[4];
    a[0].h[0] = *(const bf16x8*)(Ar0 + k +      half * 8);
    a[0].h[1] = *(const bf16x8*)(Ar0 + k + 16 + half * 8);
    a[1].h[0] = *(const bf16x8*)(Ar1 + k +      half * 8);
    a[1].h[1] = *(const bf16x8*)(Ar1 + k + 16 + half * 8);
#pragma unroll
    for (int j = 0; j < 4; ++j) {
      b[j].h[0] = *(const bf16x8*)(Br0 + (size_t)j * 16 * KDIM + k +      half * 8);
      b[j].h[1] = *(const bf16x8*)(Br0 + (size_t)j * 16 * KDIM + k + 16 + half * 8);
    }

#pragma unroll
    for (int i = 0; i < 2; ++i)
#pragma unroll
      for (int j = 0; j < 4; ++j)
        acc[i][j] = __builtin_amdgcn_wmma_f32_16x16x32_bf16(
            false, a[i].v, false, b[j].v, (short)0, acc[i][j], false, false);
  }

  // C/D layout: VGPR r -> row (r + half*8); lane l16 -> column.
#pragma unroll
  for (int i = 0; i < 2; ++i) {
#pragma unroll
    for (int j = 0; j < 4; ++j) {
      const int col = tileN0 + j * 16 + l16;
#pragma unroll
      for (int r = 0; r < 8; ++r) {
        const int row = tileM0 + i * 16 + half * 8 + r;
        if (F32OUT)
          ((float*)Cout)[(size_t)row * ldc + col] = acc[i][j][r];
        else
          ((unsigned short*)Cout)[(size_t)row * ldc + col] = f2bf(acc[i][j][r]);
      }
    }
  }
}

// ---------------------------------------------------------------------------
// Local windowed attention. One wave32 per (bn, h, s): 7 dot-products of
// length 64 (2 dims/lane + butterfly reduce), bias, softmax over the window,
// weighted sum of v. Output bf16 row-major [32768][512].
// ---------------------------------------------------------------------------
__global__ __launch_bounds__(256) void k_local_attn(
    const unsigned short* __restrict__ qkv,     // [32768][1536] bf16 (q|k|v)
    const float* __restrict__ pos_bias,         // [8][256][7] f32
    unsigned short* __restrict__ attn_out) {    // [32768][512] bf16
  const int wid  = threadIdx.x >> 5;
  const int lane = threadIdx.x & 31;
  const int g  = blockIdx.x * 8 + wid;          // 0 .. 2*64*256*8-1
  const int s  = g & (S - 1);
  const int h  = (g >> 8) & (HEADS - 1);
  const int bn = g >> 11;
  const int row = bn * S + s;
  const int d0  = lane * 2;                     // 2 dims per lane (DK=64)

  const unsigned int qp =
      *(const unsigned int*)(qkv + (size_t)row * NQKV + h * DK + d0);
  const float q0 = bf2f((unsigned short)qp);
  const float q1 = bf2f((unsigned short)(qp >> 16));

  float score[KSIZE];
#pragma unroll
  for (int w = 0; w < KSIZE; ++w) {
    const int sp = s + w - 3;
    float sc = -3.0e38f;
    if (sp >= 0 && sp < S) {
      const unsigned int kp = *(const unsigned int*)(
          qkv + (size_t)(bn * S + sp) * NQKV + 512 + h * DK + d0);
      float p = q0 * bf2f((unsigned short)kp) +
                q1 * bf2f((unsigned short)(kp >> 16));
#pragma unroll
      for (int off = 16; off > 0; off >>= 1)
        p += __shfl_xor(p, off, 32);
      sc = p + pos_bias[((h * S) + s) * KSIZE + w];
    }
    score[w] = sc;
  }

  float m = score[0];
#pragma unroll
  for (int w = 1; w < KSIZE; ++w) m = fmaxf(m, score[w]);
  float e[KSIZE], sum = 0.0f;
#pragma unroll
  for (int w = 0; w < KSIZE; ++w) { e[w] = __expf(score[w] - m); sum += e[w]; }
  const float inv = 1.0f / sum;

  float o0 = 0.0f, o1 = 0.0f;
#pragma unroll
  for (int w = 0; w < KSIZE; ++w) {
    const int sp = s + w - 3;
    if (sp >= 0 && sp < S) {
      const unsigned int vp = *(const unsigned int*)(
          qkv + (size_t)(bn * S + sp) * NQKV + 1024 + h * DK + d0);
      const float p = e[w] * inv;
      o0 += p * bf2f((unsigned short)vp);
      o1 += p * bf2f((unsigned short)(vp >> 16));
    }
  }
  const unsigned int op =
      (unsigned)f2bf(o0) | ((unsigned)f2bf(o1) << 16);
  *(unsigned int*)(attn_out + (size_t)row * 512 + h * DK + d0) = op;
}

// ---------------------------------------------------------------------------
// Launch
// ---------------------------------------------------------------------------
extern "C" void kernel_launch(void* const* d_in, const int* in_sizes, int n_in,
                              void* d_out, int out_size, void* d_ws,
                              size_t ws_size, hipStream_t stream) {
  (void)in_sizes; (void)n_in; (void)out_size; (void)ws_size;
  const float* inputs   = (const float*)d_in[0];   // [2][64][256][512]
  const float* pos_bias = (const float*)d_in[1];   // [8][256][7]
  const float* W_qkv    = (const float*)d_in[2];   // [512][1536]
  const float* W_out    = (const float*)d_in[3];   // [512][512]
  float* out = (float*)d_out;                      // [32768][512]

  uint8_t* ws = (uint8_t*)d_ws;
  // Workspace layout (all offsets 512KB-granular aligned):
  unsigned short* Xb    = (unsigned short*)(ws);                 // 32768*512  bf16 (32 MB)
  unsigned short* WqkvT = (unsigned short*)(ws + 33554432u);     // 1536*512   bf16
  unsigned short* WoutT = (unsigned short*)(ws + 35127296u);     // 512*512    bf16
  unsigned short* qkv   = (unsigned short*)(ws + 35651584u);     // 32768*1536 bf16 (96 MB)
  unsigned short* attn  = Xb;  // alias: X no longer needed after GEMM1

  // 1) inputs f32 -> bf16
  k_cvt_x<<<(MROWS * KDIM / 4 + 255) / 256, 256, 0, stream>>>(
      inputs, Xb, MROWS * KDIM / 4);

  // 2) weight transpose+convert (q columns pre-scaled by 1/sqrt(DK))
  k_prep_w<<<(NQKV * KDIM + 255) / 256, 256, 0, stream>>>(W_qkv, WqkvT, NQKV, 512);
  k_prep_w<<<(512 * KDIM + 255) / 256, 256, 0, stream>>>(W_out, WoutT, 512, 0);

  // 3) GEMM1: qkv = X @ W_qkv   (bf16 WMMA, bf16 out)
  gemm_bf16_wmma<false><<<dim3(NQKV / 128, MROWS / 128), 256, 0, stream>>>(
      Xb, WqkvT, (void*)qkv, NQKV);

  // 4) windowed softmax attention (one wave per (bn,h,s))
  k_local_attn<<<(2 * 64 * 256 * 8) / 8, 256, 0, stream>>>(qkv, pos_bias, attn);

  // 5) GEMM2: out = attn @ W_out  (bf16 WMMA, f32 out)
  gemm_bf16_wmma<true><<<dim3(512 / 128, MROWS / 128), 256, 0, stream>>>(
      attn, WoutT, (void*)out, 512);
}
